// Model_26147760898465
// MI455X (gfx1250) — matrix operand order repaired and checked by reading, not compiled, not run on hardware
//
#include <hip/hip_runtime.h>
#include <math.h>

// ---------------- model constants ----------------
#define IN_CH 1024
#define HID   256
#define T     128
#define CTX   16
#define NLAYERS 8
#define LEAK  0.2f
#define EPS_N 1e-8f

// ---------------- CDNA5 WMMA types ----------------
typedef __attribute__((ext_vector_type(16))) __bf16          v16bf;
typedef __attribute__((ext_vector_type(16))) unsigned short  v16us;
typedef __attribute__((ext_vector_type(8)))  unsigned short  v8us;
typedef __attribute__((ext_vector_type(8)))  float           v8f;

// ---------------- weight-fragment workspace layout ----------------
// A-fragments (16x32 bf16) stored in per-lane order, contiguous K (lane half h: K 16h..16h+15):
// frag[fid][lane][16] -> each lane loads 32 contiguous bytes (2x b128, coalesced).
#define PROJ_FRAGS (16 * 32)              // o_tiles x ksteps                 = 512
#define CONV_FRAGS (NLAYERS * 2 * 16 * 8) // layers x taps x o_tiles x ksteps = 2048
#define EV_FRAGS   (8)                    // ksteps
#define NFRAGS     (PROJ_FRAGS + CONV_FRAGS + EV_FRAGS)  // 2568
#define FRAG_US    (32 * 16)              // ushorts per fragment (1KB)
#define WS_X_OFF   ((size_t)4 << 20)      // x_bf16 region starts at 4MB into d_ws

// Native bf16 converts (single v_cvt, RNE) instead of integer bit tricks.
__device__ __forceinline__ unsigned short f2bf(float f) {
    __bf16 h = (__bf16)f;
    return __builtin_bit_cast(unsigned short, h);
}
__device__ __forceinline__ float bf2f(unsigned short u) {
    return (float)__builtin_bit_cast(__bf16, u);
}

__device__ __forceinline__ v8f wmma_bf16(v16bf a, v16bf b, v8f c) {
    // (neg_a, A, neg_b, B, c_mod, C, reuse_a, reuse_b)
    return __builtin_amdgcn_wmma_f32_16x16x32_bf16(false, a, false, b, (short)0, c, false, false);
}

// Pre-swizzled A fragment: 32 contiguous bytes per lane (2x global_load_b128).
__device__ __forceinline__ v16bf load_frag(const unsigned short* __restrict__ frags,
                                           int fid, int lane) {
    const v16us u = *(const v16us*)(frags + (size_t)fid * FRAG_US + lane * 16);
    return __builtin_bit_cast(v16bf, u);
}

// B fragment (32x16 bf16) from LDS activation stored time-major [T][HID]:
// lane col n holds K=k0..k0+15 (lanes 0-15) / k0+16..31 (lanes 16-31) -> contiguous.
__device__ __forceinline__ v16bf ldsB(const unsigned short* H, int k0, int col, int lane) {
    const int kb = (lane >> 4) * 16;
    v16us u;
    if (col < T) {
        u = *(const v16us*)(H + col * HID + k0 + kb);
    } else {
#pragma unroll
        for (int e = 0; e < 16; ++e) u[e] = 0;   // right (future) zero padding
    }
    return __builtin_bit_cast(v16bf, u);
}

// B fragment from pre-transposed bf16 x [T][IN_CH] in workspace.
__device__ __forceinline__ v16bf gB_bf16(const unsigned short* __restrict__ X,
                                         int k0, int col, int lane) {
    const int kb = (lane >> 4) * 16;
    const v16us u = *(const v16us*)(X + (size_t)col * IN_CH + k0 + kb);
    return __builtin_bit_cast(v16bf, u);
}

// Fallback: B fragment gathered from f32 x [IN_CH][T] (only if d_ws too small).
__device__ __forceinline__ v16bf gB_f32(const float* __restrict__ X,
                                        int k0, int col, int lane) {
    const int kb = (lane >> 4) * 16;
    v16us u;
#pragma unroll
    for (int e = 0; e < 16; ++e)
        u[e] = f2bf(X[(size_t)(k0 + kb + e) * T + col]);
    return __builtin_bit_cast(v16bf, u);
}

// Projection templated on loader so the uniform have_x branch is hoisted out of
// the hot loop (keeps the inner loop a straight 2xb128 + 4xWMMA pipeline).
template <bool HAVE_X>
__device__ __forceinline__ void do_projection(const float* __restrict__ xb,
                                              const unsigned short* __restrict__ xbfb,
                                              const unsigned short* __restrict__ frags,
                                              const float* __restrict__ proj_b,
                                              unsigned short* Hs,
                                              int col, int rb, int lane) {
#pragma unroll 1
    for (int oc = 0; oc < HID / 16; oc += 4) {
        v8f acc[4] = {};
#pragma unroll 1
        for (int ks = 0; ks < IN_CH / 32; ++ks) {
            v16bf bf = HAVE_X ? gB_bf16(xbfb, ks * 32, col, lane)
                              : gB_f32(xb, ks * 32, col, lane);
#pragma unroll
            for (int j = 0; j < 4; ++j) {
                v16bf af = load_frag(frags, (oc + j) * 32 + ks, lane);
                acc[j] = wmma_bf16(af, bf, acc[j]);
            }
        }
#pragma unroll
        for (int j = 0; j < 4; ++j) {
            v8us p;
#pragma unroll
            for (int r = 0; r < 8; ++r) {
                int o = (oc + j) * 16 + rb + r;
                p[r] = f2bf(acc[j][r] + proj_b[o]);
            }
            *(v8us*)(Hs + col * HID + (oc + j) * 16 + rb) = p;   // one ds_store_b128
        }
    }
}

// ================= prepass 1: swizzle weights into fragment order ==============
__global__ __launch_bounds__(256)
void swizzle_weights_kernel(const float* __restrict__ proj_w,
                            const float* __restrict__ conv_w,
                            const float* __restrict__ ev_w,
                            unsigned short* __restrict__ frags) {
    const int fid  = (blockIdx.x * blockDim.x + threadIdx.x) >> 5;  // one wave per frag
    const int lane = threadIdx.x & 31;
    if (fid >= NFRAGS) return;

    const float* W; int row0, rs, k0, cs, co;
    if (fid < PROJ_FRAGS) {
        W = proj_w; row0 = (fid >> 5) * 16; rs = IN_CH; k0 = (fid & 31) * 32; cs = 1; co = 0;
    } else if (fid < PROJ_FRAGS + CONV_FRAGS) {
        int c = fid - PROJ_FRAGS;
        int ks = c & 7, ot = (c >> 3) & 15, tap = (c >> 7) & 1, L = c >> 8;
        W = conv_w + (size_t)L * HID * HID * 2;
        row0 = ot * 16; rs = HID * 2; k0 = ks * 32; cs = 2; co = tap;
    } else {
        int ks = fid - PROJ_FRAGS - CONV_FRAGS;
        W = ev_w; row0 = 0; rs = HID; k0 = ks * 32; cs = 1; co = 0;
    }

    // A-layout (matches B loaders): lanes 0-15 row m=lane, K 0..15; lanes 16-31: K 16..31
    const int m  = lane & 15;
    const int kb = (lane >> 4) * 16;
    v16us r;
#pragma unroll
    for (int e = 0; e < 16; ++e) {
        int K = kb + e;
        r[e] = f2bf(W[(size_t)(row0 + m) * rs + (size_t)(k0 + K) * cs + co]);
    }
    *(v16us*)(frags + (size_t)fid * FRAG_US + lane * 16) = r;
}

// ================= prepass 2: x [B][C][T] f32 -> [B][T][C] bf16 ================
__global__ __launch_bounds__(256)
void xpose_kernel(const float* __restrict__ x, unsigned short* __restrict__ xbf) {
    __shared__ float tile[32][33];
    const int b = blockIdx.z, c0 = blockIdx.y * 32, t0 = blockIdx.x * 32;
    const int tx = threadIdx.x & 31, ty = threadIdx.x >> 5;   // ty 0..7
    const float* src = x + ((size_t)b * IN_CH + c0) * T + t0;
#pragma unroll
    for (int i = 0; i < 4; ++i)
        tile[ty + 8 * i][tx] = src[(size_t)(ty + 8 * i) * T + tx];   // coalesced read
    __syncthreads();
    unsigned short* dst = xbf + ((size_t)b * T + t0) * IN_CH + c0;
#pragma unroll
    for (int i = 0; i < 4; ++i)
        dst[(size_t)(ty + 8 * i) * IN_CH + tx] = f2bf(tile[tx][ty + 8 * i]); // coalesced write
}

// LDS: h(bf16) 64KB + y(bf16) 64KB + rnorm + ev + attn + red  (~140KB of 320KB WGP LDS)
#define SMEM_BYTES (HID*T*2 + HID*T*2 + T*4 + CTX*T*4 + T*4 + 16)

// ================= main kernel: one block (8 wave32) per batch element =========
__global__ __launch_bounds__(256)
void wavenet_main_kernel(const float* __restrict__ x,
                         const unsigned short* __restrict__ xbf, int have_x,
                         const unsigned short* __restrict__ frags,
                         const float* __restrict__ proj_b,
                         const float* __restrict__ conv_b,
                         const float* __restrict__ ev_b,
                         const float* __restrict__ sw_w, const float* __restrict__ sw_b,
                         float* __restrict__ out, int nbatch) {
    extern __shared__ char smem[];
    unsigned short* Hs    = (unsigned short*)smem;     // [T][HID] bf16, resident activation
    unsigned short* Ys    = Hs + T * HID;              // [T][HID] bf16, layer output
    float*          rnorm = (float*)(Ys + T * HID);    // [T]
    float*          evs   = rnorm + T;                 // [CTX][T]
    float*          attn  = evs + CTX * T;             // [T]
    float*          red   = attn + T;                  // [0]=val, [1]=idx bits

    const int b    = blockIdx.x;
    const int tid  = threadIdx.x;
    const int lane = tid & 31;
    const int wave = tid >> 5;          // 8 waves
    const int t0   = wave * 16;         // each wave owns one 16-wide t-tile
    const int rb   = (lane >> 4) * 8;   // C/D row base for this lane
    const int col  = t0 + (lane & 15);  // this lane's time column

    const float*          xb   = x + (size_t)b * IN_CH * T;
    const unsigned short* xbfb = xbf + (size_t)b * T * IN_CH;

    // ---------------- projection: h = proj_w[256,1024] * x ----------------
    if (have_x) do_projection<true >(xb, xbfb, frags, proj_b, Hs, col, rb, lane);
    else        do_projection<false>(xb, xbfb, frags, proj_b, Hs, col, rb, lane);
    __syncthreads();

    // ---------------- 8 dilated residual layers, h stays in LDS ----------------
    const int dil[NLAYERS] = {1, 2, 4, 8, 16, 32, 64, 1};
#pragma unroll 1
    for (int L = 0; L < NLAYERS; ++L) {
        const int d = dil[L];
        const float* Bv = conv_b + L * HID;
        const int fb0 = PROJ_FRAGS + ((L * 2 + 0) * 16) * 8;   // tap0 fragment base
        const int fb1 = PROJ_FRAGS + ((L * 2 + 1) * 16) * 8;   // tap1 fragment base

#pragma unroll 1
        for (int oc = 0; oc < HID / 16; oc += 4) {
            v8f acc[4] = {};
#pragma unroll 1
            for (int ks = 0; ks < HID / 32; ++ks) {
                v16bf b0 = ldsB(Hs, ks * 32, col, lane);        // 2x ds_load_b128
                v16bf b1 = ldsB(Hs, ks * 32, col + d, lane);    // dilated (zero-padded) tap
#pragma unroll
                for (int j = 0; j < 4; ++j) {
                    v16bf a0 = load_frag(frags, fb0 + (oc + j) * 8 + ks, lane);
                    acc[j] = wmma_bf16(a0, b0, acc[j]);
                    v16bf a1 = load_frag(frags, fb1 + (oc + j) * 8 + ks, lane);
                    acc[j] = wmma_bf16(a1, b1, acc[j]);
                }
            }
#pragma unroll
            for (int j = 0; j < 4; ++j) {
                v8us skip = *(const v8us*)(Hs + col * HID + (oc + j) * 16 + rb);
                v8us p;
#pragma unroll
                for (int r = 0; r < 8; ++r) {
                    int o = (oc + j) * 16 + rb + r;
                    float v = acc[j][r] + Bv[o];
                    v = (v > 0.0f) ? v : LEAK * v;               // leaky relu
                    v += bf2f(skip[r]);                          // residual
                    p[r] = f2bf(v);
                }
                *(v8us*)(Ys + col * HID + (oc + j) * 16 + rb) = p;
            }
        }
        __syncthreads();

        // per-timestep channel L2 norm (rows are contiguous in [T][HID] layout)
        if (tid < T) {
            float s = 0.0f;
            const unsigned short* row = Ys + tid * HID;
            for (int c = 0; c < HID; ++c) { float v = bf2f(row[c]); s += v * v; }
            rnorm[tid] = 1.0f / (sqrtf(s) + EPS_N);
        }
        __syncthreads();
        {   // h = y * rnorm : 2 threads per time row, 128 contiguous elems each
            const int t  = tid >> 1;
            const int c0 = (tid & 1) * (HID / 2);
            const float rn = rnorm[t];
            for (int c = 0; c < HID / 2; ++c)
                Hs[t * HID + c0 + c] = f2bf(bf2f(Ys[t * HID + c0 + c]) * rn);
        }
        __syncthreads();
    }

    // ---------------- event head: one 16x16 WMMA tile per t-tile ----------------
    {
        v8f acc = {};
#pragma unroll 1
        for (int ks = 0; ks < HID / 32; ++ks) {
            v16bf bf = ldsB(Hs, ks * 32, col, lane);
            v16bf af = load_frag(frags, PROJ_FRAGS + CONV_FRAGS + ks, lane);
            acc = wmma_bf16(af, bf, acc);
        }
#pragma unroll
        for (int r = 0; r < 8; ++r) {
            int c = rb + r;
            evs[c * T + col] = acc[r] + ev_b[c];
        }
    }
    // switch head + relu (contiguous channel rows)
    if (tid < T) {
        float s = sw_b[0];
        const unsigned short* row = Hs + tid * HID;
        for (int c = 0; c < HID; ++c) s += bf2f(row[c]) * sw_w[c];
        attn[tid] = (s > 0.0f) ? s : 0.0f;
    }
    __syncthreads();

    // top-1 over time (first max on ties, like lax.top_k)
    if (tid == 0) {
        float best = attn[0]; int bi = 0;
        for (int t = 1; t < T; ++t) if (attn[t] > best) { best = attn[t]; bi = t; }
        red[0] = best; ((int*)red)[1] = bi;
    }
    __syncthreads();
    const int   idx = ((int*)red)[1];
    const float val = red[0];

    // outputs: vecs [B,1,CTX] then scheduling [B,1,T], concatenated flat
    float* vecs_out  = out;                        // B*CTX
    float* sched_out = out + (size_t)nbatch * CTX; // B*T
    if (tid < CTX) vecs_out[(size_t)b * CTX + tid] = evs[tid * T + idx];
    if (tid < T)   sched_out[(size_t)b * T + tid] = (tid == idx) ? val : 0.0f;
}

extern "C" void kernel_launch(void* const* d_in, const int* in_sizes, int n_in,
                              void* d_out, int out_size, void* d_ws, size_t ws_size,
                              hipStream_t stream) {
    const float* x      = (const float*)d_in[0];
    const float* proj_w = (const float*)d_in[1];
    const float* proj_b = (const float*)d_in[2];
    const float* conv_w = (const float*)d_in[3];
    const float* conv_b = (const float*)d_in[4];
    const float* ev_w   = (const float*)d_in[5];
    const float* ev_b   = (const float*)d_in[6];
    const float* sw_w   = (const float*)d_in[7];
    const float* sw_b   = (const float*)d_in[8];
    float* out = (float*)d_out;

    const int nbatch = in_sizes[0] / (IN_CH * T);   // 256

    unsigned short* frags = (unsigned short*)d_ws;                     // 2.6MB, L2-resident
    unsigned short* xbf   = (unsigned short*)((char*)d_ws + WS_X_OFF); // 64MB bf16 x
    const size_t x_bytes  = (size_t)nbatch * T * IN_CH * sizeof(unsigned short);
    const int have_x = (ws_size >= WS_X_OFF + x_bytes) ? 1 : 0;        // constant across calls

    // prepass 1: weight fragments (one wave per fragment)
    swizzle_weights_kernel<<<NFRAGS / 8, 256, 0, stream>>>(proj_w, conv_w, ev_w, frags);

    // prepass 2: transpose+convert x (tiled via LDS; both sides coalesced)
    if (have_x)
        xpose_kernel<<<dim3(T / 32, IN_CH / 32, nbatch), 256, 0, stream>>>(x, xbf);

    // main: one workgroup per batch element, ~140KB dynamic LDS (WGP has 320KB)
    (void)hipFuncSetAttribute((const void*)wavenet_main_kernel,
                              hipFuncAttributeMaxDynamicSharedMemorySize, SMEM_BYTES);
    wavenet_main_kernel<<<nbatch, 256, SMEM_BYTES, stream>>>(
        x, xbf, have_x, frags, proj_b, conv_b, ev_b, sw_w, sw_b, out, nbatch);
}
